// Head_50766513439443
// MI455X (gfx1250) — matrix operand order repaired and checked by reading, not compiled, not run on hardware
//
#include <hip/hip_runtime.h>
#include <hip/hip_bf16.h>

// ---------------------------------------------------------------------------
// Fused multi-head-ish attention for the reference:
//   Q/K/V = x @ W{q,k,v}   (65536x256 @ 256x64, f16 storage, f32 accum WMMA)
//   per (j,b): S_i = Q[j,b] K[i,b]^T * 1/16 ; P_i = softmax_q(S_i)
//   O[j,b]   = sum_i P_i V[i,b]
// All matrix math via v_wmma_f32_16x16x32_f16 (wave32, CDNA5).
// ---------------------------------------------------------------------------

typedef _Float16 half_t;
typedef __attribute__((ext_vector_type(16))) _Float16 v16h;
typedef __attribute__((ext_vector_type(8)))  _Float16 v8h;
typedef __attribute__((ext_vector_type(8)))  float    v8f;

#define NSEQ  8
#define BATCH 64
#define HTOK  128
#define FEAT  256
#define HS    64
#define ROWS  (NSEQ * BATCH * HTOK)   // 65536

#define VT_STRIDE 136   // halves; 272 B rows -> 16B aligned, spreads banks
#define WT_STRIDE 272   // halves; 544 B rows -> 16B aligned

static __device__ __forceinline__ v8f zero8() {
  v8f z;
#pragma unroll
  for (int i = 0; i < 8; ++i) z[i] = 0.0f;
  return z;
}

static __device__ __forceinline__ v16h combine16(v8h lo, v8h hi) {
  v16h r;
#pragma unroll
  for (int i = 0; i < 8; ++i) { r[i] = lo[i]; r[i + 8] = hi[i]; }
  return r;
}

// WMMA wrapper: D = A(16x32 f16) * B(32x16 f16) + C(16x16 f32)
static __device__ __forceinline__ v8f wmma_f16(v16h a, v16h b, v8f c) {
  return __builtin_amdgcn_wmma_f32_16x16x32_f16(
      /*neg_a=*/false, a, /*neg_b=*/false, b,
      /*c_mod=*/(short)0, c, /*reuse_a=*/false, /*reuse_b=*/false);
}

// A fragment (16x32, MxK) from row-major f16 [m][k], ld in halves.
// Lane layout: m = m0 + (lane&15); halves 0..7 -> k = k0+(lane<16?0:8)+h,
// halves 8..15 -> k = k0+16+(lane<16?0:8)+(h-8).
static __device__ __forceinline__ v16h load_a_f16(const half_t* base, int ld,
                                                  int m0, int k0, int lane) {
  const int m    = m0 + (lane & 15);
  const int koff = (lane < 16) ? 0 : 8;
  const half_t* r = base + (size_t)m * ld + k0;
  v8h lo = *(const v8h*)(r + koff);
  v8h hi = *(const v8h*)(r + 16 + koff);
  return combine16(lo, hi);
}

// A fragment with on-the-fly f32 -> f16 convert (for x).
static __device__ __forceinline__ v16h load_a_f32cvt(const float* base, int ld,
                                                     int m0, int k0, int lane) {
  const int m    = m0 + (lane & 15);
  const int koff = (lane < 16) ? 0 : 8;
  const float* r = base + (size_t)m * ld + k0;
  v16h out;
#pragma unroll
  for (int i = 0; i < 8; ++i) {
    out[i]     = (_Float16)r[koff + i];
    out[i + 8] = (_Float16)r[16 + koff + i];
  }
  return out;
}

// B fragment (32x16, KxN) read from B^T stored row-major: S[n][k], ld halves.
// Lane layout: n = n0 + (lane&15); halves 0..7 -> k = k0+(lane<16?0:8)+h,
// halves 8..15 -> k = k0+16+(lane<16?0:8)+(h-8).
static __device__ __forceinline__ v16h load_bT(const half_t* S, int ld,
                                               int n0, int k0, int lane) {
  const int n    = n0 + (lane & 15);
  const int koff = (lane < 16) ? 0 : 8;
  const half_t* p = S + (size_t)n * ld + k0 + koff;
  return combine16(*(const v8h*)p, *(const v8h*)(p + 16));
}

// ---------------------------------------------------------------------------
// Kernel 1: QKV projection. Grid = ROWS/128 blocks, 256 threads (8 waves).
// Each wave owns a 16-row strip; X fragments converted to f16 once and reused
// for all three weight matrices. W^T staged in LDS per weight.
// ---------------------------------------------------------------------------
__global__ __launch_bounds__(256) void qkv_proj_kernel(
    const float* __restrict__ X, const float* __restrict__ Wq,
    const float* __restrict__ Wk, const float* __restrict__ Wv,
    half_t* __restrict__ Q, half_t* __restrict__ K, half_t* __restrict__ V) {
  __shared__ half_t Wt[HS * WT_STRIDE];  // W^T: [d][f] -> 34816 B

  const int tid  = threadIdx.x;
  const int wave = tid >> 5;
  const int lane = tid & 31;
  const size_t row0  = (size_t)blockIdx.x * 128;
  const int    pbase = wave * 16;

  const float* Ws[3]  = {Wq, Wk, Wv};
  half_t*      Out[3] = {Q, K, V};

  // X fragments for this wave's 16 rows, all 8 K-chunks (K = 256).
  const float* Xblk = X + row0 * FEAT;
  v16h ax[8];
#pragma unroll
  for (int kc = 0; kc < 8; ++kc)
    ax[kc] = load_a_f32cvt(Xblk, FEAT, pbase, kc * 32, lane);

  for (int w = 0; w < 3; ++w) {
    __syncthreads();  // prior compute done before restaging Wt
    const float* Wsrc = Ws[w];
    for (int e = tid; e < FEAT * HS; e += 256) {
      int kk = e >> 6;        // f index (coalesced global read)
      int d  = e & 63;        // output dim
      Wt[d * WT_STRIDE + kk] = (half_t)Wsrc[e];
    }
    __syncthreads();

    v8f acc[4];
#pragma unroll
    for (int n = 0; n < 4; ++n) acc[n] = zero8();

#pragma unroll
    for (int kc = 0; kc < 8; ++kc) {
#pragma unroll
      for (int n = 0; n < 4; ++n)
        acc[n] = wmma_f16(ax[kc], load_bT(Wt, WT_STRIDE, n * 16, kc * 32, lane),
                          acc[n]);
    }

    // C layout: element e -> row M = pbase + (lane<16? e : e+8), col = n*16+(lane&15)
    half_t* O = Out[w] + row0 * HS;
    const int m0 = pbase + ((lane < 16) ? 0 : 8);
#pragma unroll
    for (int n = 0; n < 4; ++n)
#pragma unroll
      for (int e = 0; e < 8; ++e)
        O[(size_t)(m0 + e) * HS + n * 16 + (lane & 15)] = (half_t)acc[n][e];
  }
}

// ---------------------------------------------------------------------------
// Kernel 2: fused attention. Grid = n*b = 512 blocks, 256 threads (8 waves).
// Block owns (j,b); wave w owns p-rows [16w,16w+16). Whole 16x128 score strip
// lives in 8 v8f accumulators; softmax via 16-lane shuffle reductions; P goes
// through per-wave LDS to be re-laid-out as an A fragment for P @ V.
// ---------------------------------------------------------------------------
__global__ __launch_bounds__(256) void attn_kernel(
    const half_t* __restrict__ Q, const half_t* __restrict__ K,
    const half_t* __restrict__ V, float* __restrict__ out) {
  __shared__ half_t Vt[HS * VT_STRIDE];       // V^T: [d][q] -> 17408 B
  __shared__ half_t Plds[8 * 16 * HTOK];      // per-wave P tile -> 32768 B

  const int blk  = blockIdx.x;
  const int j    = blk >> 6;   // 0..7
  const int b    = blk & 63;   // 0..63
  const int tid  = threadIdx.x;
  const int wave = tid >> 5;
  const int lane = tid & 31;
  const int pbase = wave * 16;
  const float scale = 0.0625f;          // f^-0.5, f = 256 (as in reference)

  const half_t* Qblk = Q + (size_t)(j * BATCH + b) * HTOK * HS;

  // Q fragments held for the whole kernel (K-dim 64 = 2 chunks).
  v16h aq0 = load_a_f16(Qblk, HS, pbase, 0, lane);
  v16h aq1 = load_a_f16(Qblk, HS, pbase, 32, lane);

  v8f o[4];
#pragma unroll
  for (int n = 0; n < 4; ++n) o[n] = zero8();

  half_t* Pw = &Plds[wave * 16 * HTOK];

  for (int i = 0; i < NSEQ; ++i) {
    const half_t* Kblk = K + (size_t)(i * BATCH + b) * HTOK * HS;
    const half_t* Vblk = V + (size_t)(i * BATCH + b) * HTOK * HS;

    // Prefetch next iteration's K/V block (global_prefetch_b8).
    if (i + 1 < NSEQ) {
      __builtin_prefetch(K + (size_t)((i + 1) * BATCH + b) * HTOK * HS + lane * 64, 0, 1);
      __builtin_prefetch(V + (size_t)((i + 1) * BATCH + b) * HTOK * HS + lane * 64, 0, 1);
    }

    // Stage V^T into LDS (coalesced global read, transposed LDS write).
    __syncthreads();  // everyone done reading previous Vt
#pragma unroll
    for (int it = 0; it < 32; ++it) {
      int e = it * 256 + tid;          // 128*64 elements
      int q = e >> 6;
      int d = e & 63;
      Vt[d * VT_STRIDE + q] = Vblk[e];
    }
    __syncthreads();

    // S strip: 16 rows x 128 cols, 8 q-tiles x 2 K-chunks of WMMA.
    v8f s[8];
#pragma unroll
    for (int t = 0; t < 8; ++t) {
      v8f acc = zero8();
      acc = wmma_f16(aq0, load_bT(Kblk, HS, t * 16, 0, lane), acc);
      acc = wmma_f16(aq1, load_bT(Kblk, HS, t * 16, 32, lane), acc);
      s[t] = acc;
    }

    // Row softmax. Element e of s[t]: row M = (lane<16? e : e+8),
    // col N = 16t + (lane&15). Reduce over t and over the 16-lane group.
#pragma unroll
    for (int e = 0; e < 8; ++e) {
      float m = -__builtin_inff();
#pragma unroll
      for (int t = 0; t < 8; ++t) m = fmaxf(m, s[t][e]);
#pragma unroll
      for (int off = 1; off < 16; off <<= 1)
        m = fmaxf(m, __shfl_xor(m, off, 32));
      float sum = 0.0f;
#pragma unroll
      for (int t = 0; t < 8; ++t) {
        float p = __expf((s[t][e] - m) * scale);
        s[t][e] = p;
        sum += p;
      }
#pragma unroll
      for (int off = 1; off < 16; off <<= 1)
        sum += __shfl_xor(sum, off, 32);
      float r = 1.0f / sum;
#pragma unroll
      for (int t = 0; t < 8; ++t) s[t][e] *= r;
    }

    // Scatter P (C layout) into per-wave LDS, row-major [m][q] f16.
    {
      const int mrow = (lane < 16) ? 0 : 8;
      const int qcol = lane & 15;
#pragma unroll
      for (int t = 0; t < 8; ++t)
#pragma unroll
        for (int e = 0; e < 8; ++e)
          Pw[(mrow + e) * HTOK + t * 16 + qcol] = (half_t)s[t][e];
    }

    // O += P @ V : K-dim q = 128 -> 4 chunks of 32; N = 64 -> 4 tiles.
#pragma unroll
    for (int kc = 0; kc < 4; ++kc) {
      v16h ap = load_a_f16(Pw, HTOK, 0, kc * 32, lane);
#pragma unroll
      for (int n = 0; n < 4; ++n)
        o[n] = wmma_f16(ap, load_bT(Vt, VT_STRIDE, n * 16, kc * 32, lane), o[n]);
    }
  }

  // Write O (f32). C layout as above.
  float* Oblk = out + (size_t)(j * BATCH + b) * HTOK * HS;
  const int m0 = pbase + ((lane < 16) ? 0 : 8);
#pragma unroll
  for (int n = 0; n < 4; ++n)
#pragma unroll
    for (int e = 0; e < 8; ++e)
      Oblk[(size_t)(m0 + e) * HS + n * 16 + (lane & 15)] = o[n][e];
}

// ---------------------------------------------------------------------------
extern "C" void kernel_launch(void* const* d_in, const int* in_sizes, int n_in,
                              void* d_out, int out_size, void* d_ws, size_t ws_size,
                              hipStream_t stream) {
  const float* x  = (const float*)d_in[0];  // (8,64,128,256)
  const float* Wq = (const float*)d_in[1];  // (256,64)
  const float* Wk = (const float*)d_in[2];
  const float* Wv = (const float*)d_in[3];
  float* out = (float*)d_out;               // (8,64,128,64)

  // Workspace: Q/K/V in f16, 3 * 65536 * 64 * 2 = 24 MiB.
  half_t* Qw = (half_t*)d_ws;
  half_t* Kw = Qw + (size_t)ROWS * HS;
  half_t* Vw = Kw + (size_t)ROWS * HS;

  qkv_proj_kernel<<<ROWS / 128, 256, 0, stream>>>(x, Wq, Wk, Wv, Qw, Kw, Vw);
  attn_kernel<<<NSEQ * BATCH, 256, 0, stream>>>(Qw, Kw, Vw, out);
}
